// Head_12163347383051
// MI455X (gfx1250) — matrix operand order repaired and checked by reading, not compiled, not run on hardware
//
#include <hip/hip_runtime.h>
#include <hip/hip_bf16.h>

// Problem constants (from reference setup_inputs)
#define BATCH 8
#define SEQ   2048
#define CDIM  1024
#define HDIM  64

typedef __bf16 v16bf __attribute__((ext_vector_type(16)));
typedef float  v8f   __attribute__((ext_vector_type(8)));
typedef int    v4i   __attribute__((ext_vector_type(4)));

typedef __attribute__((address_space(1))) v4i* gv4i_p;  // global int4*
typedef __attribute__((address_space(3))) v4i* lv4i_p;  // LDS int4*

#define LOG2E 1.4426950408889634f

#if __has_builtin(__builtin_amdgcn_global_load_async_to_lds_b128)
#define HAVE_ASYNC_LDS 1
#else
#define HAVE_ASYNC_LDS 0
#endif

__device__ __forceinline__ v8f wmma_bf16(v16bf a, v16bf b, v8f c) {
  // D = A(16x32 bf16) * B(32x16 bf16) + C(16x16 f32)
  return __builtin_amdgcn_wmma_f32_16x16x32_bf16(
      /*neg_a=*/false, a, /*neg_b=*/false, b,
      /*c_mod=*/(short)0, c, /*reuse_a=*/false, /*reuse_b=*/false);
}

// 16-byte copy global->LDS: CDNA5 async path when available (ASYNCcnt),
// else VGPR bounce.
__device__ __forceinline__ void cp16_g2l(const void* g, void* l) {
#if HAVE_ASYNC_LDS
  __builtin_amdgcn_global_load_async_to_lds_b128((gv4i_p)g, (lv4i_p)l, 0, 0);
#else
  *(uint4*)l = *(const uint4*)g;
#endif
}

__device__ __forceinline__ void cp_wait() {
#if HAVE_ASYNC_LDS
#if __has_builtin(__builtin_amdgcn_s_wait_asynccnt)
  __builtin_amdgcn_s_wait_asynccnt(0);
#else
  asm volatile("s_wait_asynccnt 0" ::: "memory");
#endif
#endif
}

// ---- Fragment loaders (layouts per CDNA5 ISA 7.12.2, wave32) ----
// A 16x32 bf16: lane<16 -> row M=lane, K in {0..7,16..23}; lane>=16 -> K+8.
// VGPR v packs K = (v<4 ? 2v : 2v+8), +1 (low/high half of dword).
__device__ __forceinline__ v16bf load_a_bf16(const __bf16* __restrict__ base, int ld) {
  const int lane = threadIdx.x & 31;
  const int m = lane & 15;
  const int hk = (lane >= 16) ? 8 : 0;
  const __bf16* r = base + (size_t)m * ld + hk;
  v16bf f;
#pragma unroll
  for (int v = 0; v < 8; ++v) {
    int kk = (v < 4) ? (2 * v) : (2 * v + 8);
    f[2 * v]     = r[kk];
    f[2 * v + 1] = r[kk + 1];
  }
  return f;
}

__device__ __forceinline__ v16bf load_a_f32(const float* __restrict__ base, int ld) {
  const int lane = threadIdx.x & 31;
  const int m = lane & 15;
  const int hk = (lane >= 16) ? 8 : 0;
  const float* r = base + (size_t)m * ld + hk;
  v16bf f;
#pragma unroll
  for (int v = 0; v < 8; ++v) {
    int kk = (v < 4) ? (2 * v) : (2 * v + 8);
    f[2 * v]     = (__bf16)r[kk];
    f[2 * v + 1] = (__bf16)r[kk + 1];
  }
  return f;
}

// B 32x16 bf16: lane%16 -> column N; lanes0-15 hold K={0..7,16..23}, lanes16-31 K+8,
// matching the A layout. Source: element(k,n) at base + n*ldn + k (k contiguous),
// so each lane reads two runs of 8 contiguous bf16.
__device__ __forceinline__ v16bf load_b_bf16(const __bf16* __restrict__ base, int ldn) {
  const int lane = threadIdx.x & 31;
  const int n = lane & 15;
  const int hk = (lane >= 16) ? 8 : 0;
  const __bf16* r = base + (size_t)n * ldn + hk;
  v16bf f;
#pragma unroll
  for (int i = 0; i < 16; ++i) f[i] = r[(i < 8) ? i : (i + 8)];
  return f;
}

// 16-lane row reductions (rows live in lane halves under wave32 D layout)
__device__ __forceinline__ float rowmax16(float v) {
#pragma unroll
  for (int m = 1; m <= 8; m <<= 1) v = fmaxf(v, __shfl_xor(v, m, 32));
  return v;
}
__device__ __forceinline__ float rowsum16(float v) {
#pragma unroll
  for (int m = 1; m <= 8; m <<= 1) v += __shfl_xor(v, m, 32);
  return v;
}

// ============================================================
// Kernel 0: one-shot W transpose+convert: Wt[3][64][1024] bf16, Wt[s][h][c]=W_s[c][h]
// ============================================================
__global__ __launch_bounds__(256) void wt_kernel(
    const float* __restrict__ Wq, const float* __restrict__ Wk,
    const float* __restrict__ Wv, __bf16* __restrict__ Wt) {
  int id = blockIdx.x * 256 + threadIdx.x;           // 0 .. 3*64*1024-1
  int s = id / (HDIM * CDIM);
  int r = id % (HDIM * CDIM);
  int h = r / CDIM;
  int c = r % CDIM;
  const float* W = (s == 0) ? Wq : (s == 1) ? Wk : Wv;
  Wt[id] = (__bf16)W[(size_t)c * HDIM + h];
}

// ============================================================
// Kernel 1: fused QKV projection. grid=BT/64, block=128 (4 waves).
// Each wave: 16 rows; one x A-fragment feeds 12 WMMAs (3 outputs x 4 N-tiles).
// Q,K bf16 row-major [B*T][64]; V stored transposed: Vt bf16 [B][64][T].
// ============================================================
__global__ __launch_bounds__(128) void qkv_proj_kernel(
    const float* __restrict__ x, const __bf16* __restrict__ Wt,
    __bf16* __restrict__ Qb, __bf16* __restrict__ Kb, __bf16* __restrict__ Vtb) {
  const int wave = threadIdx.x >> 5;
  const int lane = threadIdx.x & 31;
  const int n = lane & 15;
  const int hrow = (lane >= 16) ? 8 : 0;
  const int r0 = blockIdx.x * 64 + wave * 16;

  v8f acc[3][4] = {};

  for (int k0 = 0; k0 < CDIM; k0 += 32) {
    v16bf a = load_a_f32(x + (size_t)r0 * CDIM + k0, CDIM);
#pragma unroll
    for (int s = 0; s < 3; ++s)
#pragma unroll
      for (int nt = 0; nt < 4; ++nt) {
        v16bf b = load_b_bf16(
            Wt + (size_t)s * HDIM * CDIM + (size_t)(nt * 16) * CDIM + k0, CDIM);
        acc[s][nt] = wmma_bf16(a, b, acc[s][nt]);
      }
  }

  // Q, K row-major
#pragma unroll
  for (int s = 0; s < 2; ++s) {
    __bf16* O = (s == 0) ? Qb : Kb;
#pragma unroll
    for (int nt = 0; nt < 4; ++nt)
#pragma unroll
      for (int j = 0; j < 8; ++j)
        O[(size_t)(r0 + j + hrow) * HDIM + nt * 16 + n] = (__bf16)acc[s][nt][j];
  }
  // V transposed: lane stores 8 contiguous bf16 along T
  {
    const int bb = r0 >> 11;        // r0 / SEQ
    const int t0 = r0 & (SEQ - 1);  // r0 % SEQ
#pragma unroll
    for (int nt = 0; nt < 4; ++nt) {
      const int h = nt * 16 + n;
      __bf16* base = Vtb + (size_t)bb * HDIM * SEQ + (size_t)h * SEQ + t0 + hrow;
#pragma unroll
      for (int j = 0; j < 8; ++j) base[j] = (__bf16)acc[2][nt][j];
    }
  }
}

// ============================================================
// Kernel 2: causal flash attention. grid = B*T/64 blocks, block=128 (4 waves).
// Wave w owns query rows [qbase + 16w, +16). KV tiles of 64 rows staged in LDS
// via async-to-LDS. Uniform trip count per block (to diagonal end) so
// __syncthreads is legal; overshoot tiles fully mask to -inf (p=0, no-op).
// ============================================================
__global__ __launch_bounds__(128) void attn_kernel(
    const __bf16* __restrict__ Qb, const __bf16* __restrict__ Kb,
    const __bf16* __restrict__ Vtb, float* __restrict__ out) {
  __shared__ __bf16 Klds[64 * 64];      // [kvrow][h]
  __shared__ __bf16 Vlds[64 * 64];      // [h][kvrow]  (transposed V)
  __shared__ __bf16 Plds[4 * 16 * 64];  // per-wave P scratch [16][64]

  const int tid  = threadIdx.x;
  const int wave = tid >> 5;
  const int lane = tid & 31;
  const int n = lane & 15;
  const int hrow = (lane >= 16) ? 8 : 0;

  const int tilesPerBatch = SEQ / 64;
  const int bb    = blockIdx.x / tilesPerBatch;
  const int qbase = (blockIdx.x % tilesPerBatch) * 64;
  const int q0    = qbase + wave * 16;

  // Q fragments for this wave's 16 rows (contraction over H=64 -> 2 k-steps)
  const __bf16* qptr = Qb + (size_t)(bb * SEQ + q0) * HDIM;
  v16bf aQ0 = load_a_bf16(qptr, HDIM);
  v16bf aQ1 = load_a_bf16(qptr + 32, HDIM);

  v8f acc[4] = {};
  float mrow[8], lrow[8];
#pragma unroll
  for (int j = 0; j < 8; ++j) { mrow[j] = -__builtin_inff(); lrow[j] = 0.0f; }

  const float scale = 0.125f;   // 1/sqrt(64)
  const int kend = qbase + 64;  // uniform across the block (causal bound)

  for (int kv0 = 0; kv0 < kend; kv0 += 64) {
    // ---- stage K tile [64][64] and Vt tile [64][64] into LDS ----
    {
      const __bf16* gk = Kb + (size_t)(bb * SEQ + kv0) * HDIM;
      const __bf16* gv = Vtb + (size_t)bb * HDIM * SEQ + kv0;
#pragma unroll
      for (int i = 0; i < 4; ++i) {
        int idx = tid + i * 128;  // 0..511 chunks of 16B
        cp16_g2l(gk + (size_t)idx * 8, Klds + (size_t)idx * 8);
        int h = idx >> 3, c = idx & 7;
        cp16_g2l(gv + (size_t)h * SEQ + c * 8, Vlds + h * 64 + c * 8);
      }
      cp_wait();
    }
    __syncthreads();

    // ---- S = Q * K^T (16x64 per wave) ----
    v8f s[4] = {};
#pragma unroll
    for (int nt = 0; nt < 4; ++nt) {
      s[nt] = wmma_bf16(aQ0, load_b_bf16(Klds + (nt * 16) * 64 + 0, 64), s[nt]);
      s[nt] = wmma_bf16(aQ1, load_b_bf16(Klds + (nt * 16) * 64 + 32, 64), s[nt]);
    }

    // ---- scale + causal mask ----
#pragma unroll
    for (int nt = 0; nt < 4; ++nt) {
      const int col = kv0 + nt * 16 + n;
#pragma unroll
      for (int j = 0; j < 8; ++j) {
        float v = s[nt][j] * scale;
        const int row = q0 + j + hrow;
        s[nt][j] = (col > row) ? -__builtin_inff() : v;
      }
    }

    // ---- online softmax ----
    float csc[8];
#pragma unroll
    for (int j = 0; j < 8; ++j) {
      float mj = fmaxf(fmaxf(s[0][j], s[1][j]), fmaxf(s[2][j], s[3][j]));
      mj = rowmax16(mj);
      float mnew = fmaxf(mrow[j], mj);
      csc[j] = exp2f((mrow[j] - mnew) * LOG2E);
      mrow[j] = mnew;
    }
#pragma unroll
    for (int nt = 0; nt < 4; ++nt)
#pragma unroll
      for (int j = 0; j < 8; ++j)
        s[nt][j] = exp2f((s[nt][j] - mrow[j]) * LOG2E);
#pragma unroll
    for (int j = 0; j < 8; ++j) {
      float rs = (s[0][j] + s[1][j]) + (s[2][j] + s[3][j]);
      rs = rowsum16(rs);
      lrow[j] = lrow[j] * csc[j] + rs;
    }
#pragma unroll
    for (int nt = 0; nt < 4; ++nt)
#pragma unroll
      for (int j = 0; j < 8; ++j) acc[nt][j] *= csc[j];

    // ---- P (D layout) -> LDS -> A layout; O += P * V ----
    __bf16* pl = Plds + wave * (16 * 64);
#pragma unroll
    for (int nt = 0; nt < 4; ++nt)
#pragma unroll
      for (int j = 0; j < 8; ++j)
        pl[(j + hrow) * 64 + nt * 16 + n] = (__bf16)s[nt][j];

    // per-wave LDS ops are in-order; fence keeps compiler from reordering
    asm volatile("s_wait_dscnt 0" ::: "memory");

    v16bf aP0 = load_a_bf16(pl, 64);
    v16bf aP1 = load_a_bf16(pl + 32, 64);
#pragma unroll
    for (int nt = 0; nt < 4; ++nt) {
      acc[nt] = wmma_bf16(aP0, load_b_bf16(Vlds + (nt * 16) * 64 + 0, 64), acc[nt]);
      acc[nt] = wmma_bf16(aP1, load_b_bf16(Vlds + (nt * 16) * 64 + 32, 64), acc[nt]);
    }

    __syncthreads();  // protect Klds/Vlds before next iteration's staging
  }

  // ---- normalize and store fp32 output [B][T][H] ----
  float invl[8];
#pragma unroll
  for (int j = 0; j < 8; ++j) invl[j] = 1.0f / lrow[j];
#pragma unroll
  for (int nt = 0; nt < 4; ++nt)
#pragma unroll
    for (int j = 0; j < 8; ++j)
      out[(size_t)(bb * SEQ + q0 + j + hrow) * HDIM + nt * 16 + n] =
          acc[nt][j] * invl[j];
}

// ============================================================
extern "C" void kernel_launch(void* const* d_in, const int* in_sizes, int n_in,
                              void* d_out, int out_size, void* d_ws, size_t ws_size,
                              hipStream_t stream) {
  const float* x  = (const float*)d_in[0];
  const float* Wq = (const float*)d_in[1];
  const float* Wk = (const float*)d_in[2];
  const float* Wv = (const float*)d_in[3];
  float* out = (float*)d_out;

  const size_t nQKV = (size_t)BATCH * SEQ * HDIM;  // 1,048,576 elems
  __bf16* Qb  = (__bf16*)d_ws;
  __bf16* Kb  = Qb + nQKV;
  __bf16* Vtb = Kb + nQKV;
  __bf16* Wt  = Vtb + nQKV;  // 3*64*1024 bf16 = 384 KB; total ws ~6.4 MB

  wt_kernel<<<dim3((3 * HDIM * CDIM) / 256), dim3(256), 0, stream>>>(Wq, Wk, Wv, Wt);

  qkv_proj_kernel<<<dim3((BATCH * SEQ) / 64), dim3(128), 0, stream>>>(
      x, Wt, Qb, Kb, Vtb);

  attn_kernel<<<dim3((BATCH * SEQ) / 64), dim3(128), 0, stream>>>(Qb, Kb, Vtb, out);
}
